// ECG_XNOR_Full_Bin_BinaryInput_5007931867428
// MI455X (gfx1250) — 8-bit matrix operand order repaired and checked by reading, compiled, not run on hardware
//
#include <hip/hip_runtime.h>

typedef __attribute__((ext_vector_type(8))) int v8i;

#define BN_EPS 1e-5f

typedef int v4i_t __attribute__((vector_size(16)));
typedef int v2i_t __attribute__((vector_size(8)));
typedef __attribute__((address_space(1))) v4i_t as1_v4i;
typedef __attribute__((address_space(3))) v4i_t as3_v4i;
typedef __attribute__((address_space(1))) v2i_t as1_v2i;
typedef __attribute__((address_space(3))) v2i_t as3_v2i;

#if __has_builtin(__builtin_amdgcn_global_load_async_to_lds_b128)
#define HAS_ASYNC128 1
#else
#define HAS_ASYNC128 0
#endif
#if __has_builtin(__builtin_amdgcn_global_load_async_to_lds_b64)
#define HAS_ASYNC64 1
#else
#define HAS_ASYNC64 0
#endif

__device__ __forceinline__ void s_wait_async(int n) {
#if __has_builtin(__builtin_amdgcn_s_wait_asynccnt)
  if (n == 0) __builtin_amdgcn_s_wait_asynccnt(0);
  else        __builtin_amdgcn_s_wait_asynccnt(2);
#else
#if HAS_ASYNC128 || HAS_ASYNC64
  asm volatile("s_wait_asynccnt 0x0" ::: "memory");
#endif
  (void)n;
#endif
}

__device__ __forceinline__ void cp16(const signed char* src, signed char* dst) {
#if HAS_ASYNC128
  __builtin_amdgcn_global_load_async_to_lds_b128(
      (as1_v4i*)(const void*)src, (as3_v4i*)(void*)dst, 0, 0);
#else
  *(int4*)dst = *(const int4*)src;
#endif
}

__device__ __forceinline__ void cp8(const signed char* src, signed char* dst) {
#if HAS_ASYNC64
  __builtin_amdgcn_global_load_async_to_lds_b64(
      (as1_v2i*)(const void*)src, (as3_v2i*)(void*)dst, 0, 0);
#else
  *(unsigned long long*)dst = *(const unsigned long long*)src;
#endif
}

__global__ void prepack_wpad(const float* __restrict__ w, signed char* __restrict__ dst,
                             int Cout, int log2c, int KPAD, int n) {
  int i = blockIdx.x * blockDim.x + threadIdx.x;
  if (i >= n) return;
  int k    = i % KPAD;
  int r    = (i / KPAD) & 15;
  int tile = i / (KPAD * 16);
  int co   = tile * 16 + r;
  int CIN  = 1 << log2c;
  signed char val = 0;
  if (co < Cout && k < 7 * CIN) {
    int dt = k >> log2c;
    int ci = k & (CIN - 1);
    val = (w[(co * CIN + ci) * 7 + dt] >= 0.f) ? (signed char)1 : (signed char)-1;
  }
  dst[i] = val;
}

__global__ void bin_input(const float* __restrict__ x, signed char* __restrict__ act, int n,
                          const float* __restrict__ g, const float* __restrict__ b,
                          const float* __restrict__ m, const float* __restrict__ v) {
  int i = blockIdx.x * blockDim.x + threadIdx.x;
  if (i >= n) return;
  float s  = g[0] * rsqrtf(v[0] + BN_EPS);
  float xb = (x[i] >= 0.f) ? 1.f : -1.f;
  float y  = (xb - m[0]) * s + b[0];
  act[i] = (y >= 0.f) ? (signed char)1 : (signed char)-1;
}

template<int CIN>
__device__ __forceinline__ void stage_a(const signed char* actBase, int Lin, int t0,
                                        int kt, signed char* bufA, signed char* trash,
                                        int lane) {
  if constexpr (CIN >= 16) {
    #pragma unroll
    for (int h = 0; h < 2; ++h) {
      int p   = lane + h * 32;
      int row = p >> 2;
      int q   = p & 3;
      int k0  = kt * 64 + q * 16;
      int dt  = k0 / CIN;
      int ci0 = k0 % CIN;
      int t   = t0 + row + dt - 3;
      bool ok = (dt < 7) && (t >= 0) && (t < Lin);
      signed char*       dst = bufA + row * 64 + q * 16;
      const signed char* src = actBase + (size_t)t * CIN + ci0;
#if HAS_ASYNC128
      cp16(ok ? src : actBase, ok ? dst : trash);
      if (!ok) { int4 z = {0, 0, 0, 0}; *(int4*)dst = z; }
#else
      int4 z = {0, 0, 0, 0};
      *(int4*)dst = ok ? *(const int4*)src : z;
#endif
    }
  } else if constexpr (CIN == 8) {
    #pragma unroll
    for (int h = 0; h < 4; ++h) {
      int p   = lane + h * 32;
      int row = p >> 3;
      int q   = p & 7;
      int t   = t0 + row + q - 3;
      bool ok = (q < 7) && (t >= 0) && (t < Lin);
      signed char*       dst = bufA + row * 64 + q * 8;
      const signed char* src = actBase + (size_t)t * 8;
#if HAS_ASYNC64
      cp8(ok ? src : actBase, ok ? dst : trash);
      if (!ok) *(unsigned long long*)dst = 0ull;
#else
      *(unsigned long long*)dst = ok ? *(const unsigned long long*)src : 0ull;
#endif
    }
  } else {
    for (int j = 0; j < 32; ++j) {
      int idx = lane * 32 + j;
      int row = idx >> 6;
      int off = idx & 63;
      int t   = t0 + row + off - 3;
      signed char v = 0;
      if (off < 7 && t >= 0 && t < Lin) v = actBase[t];
      bufA[idx] = v;
    }
  }
}

template<int CIN>
__global__ __launch_bounds__(32)
void conv_pool_wmma(const signed char* __restrict__ actIn, int Lin,
                    const signed char* __restrict__ wpad, int Cout, int Lpool,
                    signed char* __restrict__ actOut, int* __restrict__ yOut,
                    const float* __restrict__ gN, const float* __restrict__ bN,
                    const float* __restrict__ mN, const float* __restrict__ vN) {
  constexpr int K    = CIN * 7;
  constexpr int KT   = (K + 63) / 64;
  constexpr int KPAD = KT * 64;

  __shared__ __align__(16) signed char ldsA[2][16 * 64];
  __shared__ __align__(16) signed char ldsB[16 * KPAD];
  __shared__ __align__(16) int         ldsC[16 * 16];
  __shared__ __align__(16) signed char trash[16];

  const int lane = threadIdx.x;
  const int half = lane >> 4;
  const int bidx = blockIdx.z;
  const int n0   = blockIdx.y << 4;
  const int tp0  = blockIdx.x * 5;
  const int t0   = tp0 * 2;

  const signed char* actBase = actIn + (size_t)bidx * Lin * CIN;

  {
    const signed char* src = wpad + (size_t)blockIdx.y * 16 * KPAD;
    #pragma unroll
    for (int ofs = 0; ofs < 16 * KPAD; ofs += 512)
      cp16(src + ofs + lane * 16, &ldsB[ofs + lane * 16]);
  }
  stage_a<CIN>(actBase, Lin, t0, 0, ldsA[0], trash, lane);

  v8i acc = {};
  #pragma unroll
  for (int kt = 0; kt < KT; ++kt) {
    if (kt + 1 < KT)
      stage_a<CIN>(actBase, Lin, t0, kt + 1, ldsA[(kt + 1) & 1], trash, lane);
    s_wait_async((kt + 1 < KT) ? 2 : 0);
    __syncthreads();

    const int* la = (const int*)ldsA[kt & 1];
    const int* lb = (const int*)ldsB;
    int aoff = (lane & 15) * 16 + half * 2;
    v8i afrag, bfrag;
    afrag[0] = la[aoff + 0];  afrag[1] = la[aoff + 1];
    afrag[2] = la[aoff + 4];  afrag[3] = la[aoff + 5];
    afrag[4] = la[aoff + 8];  afrag[5] = la[aoff + 9];
    afrag[6] = la[aoff + 12]; afrag[7] = la[aoff + 13];
    int boff = (lane & 15) * (KPAD / 4) + kt * 16 + half * 2;
    bfrag[0] = lb[boff + 0];  bfrag[1] = lb[boff + 1];
    bfrag[2] = lb[boff + 4];  bfrag[3] = lb[boff + 5];
    bfrag[4] = lb[boff + 8];  bfrag[5] = lb[boff + 9];
    bfrag[6] = lb[boff + 12]; bfrag[7] = lb[boff + 13];

    acc = __builtin_amdgcn_wmma_i32_16x16x64_iu8(true, afrag, true, bfrag, acc,
                                                 false, false);
  }

  __syncthreads();
  {
    int nc = lane & 15;
    #pragma unroll
    for (int r = 0; r < 8; ++r) ldsC[(r + half * 8) * 16 + nc] = acc[r];
  }
  __syncthreads();

  for (int it = 0; it < 3; ++it) {
    int task = lane + it * 32;
    if (task < 80) {
      int tpl = task >> 4;
      int nc  = task & 15;
      int tp  = tp0 + tpl;
      int c   = n0 + nc;
      if (tp < Lpool && c < Cout) {
        int mv = ldsC[(2 * tpl) * 16 + nc];
        #pragma unroll
        for (int j = 1; j < 7; ++j) {
          int v2 = ldsC[(2 * tpl + j) * 16 + nc];
          mv = (v2 > mv) ? v2 : mv;
        }
        size_t o = ((size_t)bidx * Lpool + tp) * Cout + c;
        if (actOut) {
          float s  = gN[c] * rsqrtf(vN[c] + BN_EPS);
          float th = mN[c] - bN[c] / s;
          actOut[o] = ((float)mv >= th) ? (signed char)1 : (signed char)-1;
        } else {
          yOut[o] = mv;
        }
      }
    }
  }
}

__global__ void mean_out(const int* __restrict__ y7, float* __restrict__ out, int n) {
  int i = blockIdx.x * blockDim.x + threadIdx.x;
  if (i >= n) return;
  int b = i / 5, c = i % 5;
  float s = 0.f;
  #pragma unroll
  for (int j = 0; j < 23; ++j) s += (float)y7[(b * 23 + j) * 5 + c];
  out[i] = s / 23.0f;
}

extern "C" void kernel_launch(void* const* d_in, const int* in_sizes, int n_in,
                              void* d_out, int out_size, void* d_ws, size_t ws_size,
                              hipStream_t stream) {
  (void)in_sizes; (void)n_in; (void)out_size; (void)ws_size;
  const int B = 4096;
  static const int CINV[7]  = {1, 8, 16, 32, 32, 64, 64};
  static const int LOG2C[7] = {0, 3, 4, 5, 5, 6, 6};
  static const int COUT[7]  = {8, 16, 32, 32, 64, 64, 5};
  static const int LIN[7]   = {3600, 1797, 896, 445, 220, 107, 51};
  static const int LPOOL[7] = {1797, 896, 445, 220, 107, 51, 23};
  static const int KPADV[7] = {64, 64, 128, 256, 256, 448, 448};
  static const int NT[7]    = {1, 1, 2, 2, 4, 4, 1};

  char* ws = (char*)d_ws;
  size_t off = 0;
  auto carve = [&](size_t bytes) -> char* {
    char* p = ws + off;
    off += (bytes + 255) & ~(size_t)255;
    return p;
  };
  const size_t ACT_BYTES = (size_t)B * 8 * 1797;
  signed char* actPing = (signed char*)carve(ACT_BYTES);
  signed char* actPong = (signed char*)carve(ACT_BYTES);
  signed char* wp[7];
  for (int i = 0; i < 7; ++i) wp[i] = (signed char*)carve((size_t)NT[i] * 16 * KPADV[i]);
  int* y7 = (int*)carve((size_t)B * 23 * 5 * sizeof(int));

  for (int i = 0; i < 7; ++i) {
    int n = NT[i] * 16 * KPADV[i];
    prepack_wpad<<<(n + 255) / 256, 256, 0, stream>>>(
        (const float*)d_in[5 * i + 1], wp[i], COUT[i], LOG2C[i], KPADV[i], n);
  }
  {
    int n = B * 3600;
    bin_input<<<(n + 255) / 256, 256, 0, stream>>>(
        (const float*)d_in[0], actPing, n,
        (const float*)d_in[2], (const float*)d_in[3],
        (const float*)d_in[4], (const float*)d_in[5]);
  }
  signed char* cur = actPing;
  signed char* nxt = actPong;
  for (int i = 0; i < 7; ++i) {
    dim3 grid((LPOOL[i] + 4) / 5, (COUT[i] + 15) / 16, B);
    signed char* aout = (i < 6) ? nxt : nullptr;
    int*         yout = (i < 6) ? nullptr : y7;
    const float* g_ = (i < 6) ? (const float*)d_in[5 * (i + 1) + 2] : nullptr;
    const float* b_ = (i < 6) ? (const float*)d_in[5 * (i + 1) + 3] : nullptr;
    const float* m_ = (i < 6) ? (const float*)d_in[5 * (i + 1) + 4] : nullptr;
    const float* v_ = (i < 6) ? (const float*)d_in[5 * (i + 1) + 5] : nullptr;
    switch (CINV[i]) {
      case 1:  conv_pool_wmma<1 ><<<grid, 32, 0, stream>>>(cur, LIN[i], wp[i], COUT[i], LPOOL[i], aout, yout, g_, b_, m_, v_); break;
      case 8:  conv_pool_wmma<8 ><<<grid, 32, 0, stream>>>(cur, LIN[i], wp[i], COUT[i], LPOOL[i], aout, yout, g_, b_, m_, v_); break;
      case 16: conv_pool_wmma<16><<<grid, 32, 0, stream>>>(cur, LIN[i], wp[i], COUT[i], LPOOL[i], aout, yout, g_, b_, m_, v_); break;
      case 32: conv_pool_wmma<32><<<grid, 32, 0, stream>>>(cur, LIN[i], wp[i], COUT[i], LPOOL[i], aout, yout, g_, b_, m_, v_); break;
      default: conv_pool_wmma<64><<<grid, 32, 0, stream>>>(cur, LIN[i], wp[i], COUT[i], LPOOL[i], aout, yout, g_, b_, m_, v_); break;
    }
    if (i < 6) { signed char* t = cur; cur = nxt; nxt = t; }
  }
  {
    int n = B * 5;
    mean_out<<<(n + 255) / 256, 256, 0, stream>>>(y7, (float*)d_out, n);
  }
}
